// CrossScaleAttention_88433376625161
// MI455X (gfx1250) — matrix operand order repaired and checked by reading, not compiled, not run on hardware
//
#include <hip/hip_runtime.h>

// ---------------------------------------------------------------------------
// CrossScaleAttention for MI455X (gfx1250, wave32, WMMA bf16 16x16x32)
// B=4, Nq=Nk=4096, D=256, scale = 32^-0.5, L2-normalized Q/K, softmax, PV.
// Double-buffered LDS tiles fed by CDNA5 async-to-LDS loads (ASYNCcnt).
// ---------------------------------------------------------------------------

typedef __attribute__((ext_vector_type(16))) __bf16 v16bf;
typedef __attribute__((ext_vector_type(8)))  __bf16 v8bf;
typedef __attribute__((ext_vector_type(8)))  float  v8f;
typedef __attribute__((ext_vector_type(4)))  float  v4f;

#define BATCH   4
#define NQ      4096
#define NK      4096
#define DDIM    256
#define BM      64          // query rows per workgroup (4 waves x 16)
#define BN      64          // key rows per tile
#define KSTRIDE 264         // 256 + 8 bf16 pad -> 528 B rows (16B aligned, bank-spread)
#define VSTRIDE 72          // 64  + 8 bf16 pad -> 144 B rows (16B aligned, bank-spread)
#define PSTRIDE 72          // P staging stride
#define SCALE   0.17677669529663687f   // 32^-0.5

static __device__ __forceinline__ __bf16 to_bf16(float f) {
    unsigned u = __float_as_uint(f);
    u += 0x7FFFu + ((u >> 16) & 1u);          // round-to-nearest-even
    unsigned short s = (unsigned short)(u >> 16);
    return __builtin_bit_cast(__bf16, s);
}

static __device__ __forceinline__ v16bf cat8(v8bf lo, v8bf hi) {
    return __builtin_shufflevector(lo, hi, 0,1,2,3,4,5,6,7,8,9,10,11,12,13,14,15);
}

static __device__ __forceinline__ float rmax16(float v) {   // reduce within 16-lane half
    v = fmaxf(v, __shfl_xor(v, 1, 32));
    v = fmaxf(v, __shfl_xor(v, 2, 32));
    v = fmaxf(v, __shfl_xor(v, 4, 32));
    v = fmaxf(v, __shfl_xor(v, 8, 32));
    return v;
}
static __device__ __forceinline__ float rsum16(float v) {
    v += __shfl_xor(v, 1, 32);
    v += __shfl_xor(v, 2, 32);
    v += __shfl_xor(v, 4, 32);
    v += __shfl_xor(v, 8, 32);
    return v;
}

// CDNA5 async copy: per-lane 16B, global -> LDS, tracked by ASYNCcnt.
// GVS form: vdst = 32-bit LDS byte address, vaddr = 32-bit byte offset,
// saddr = 64-bit base; offset immediate applies to BOTH LDS and memory addr.
#define ASYNC_LD16(L, V, S, IMM)                                            \
    asm volatile("global_load_async_to_lds_b128 %0, %1, %2 offset:" #IMM    \
                 :: "v"(L), "v"(V), "s"(S) : "memory")

#define ASYNC_TILE16(L, V, S)                                               \
    ASYNC_LD16(L, V, S, 0);   ASYNC_LD16(L, V, S, 16);                      \
    ASYNC_LD16(L, V, S, 32);  ASYNC_LD16(L, V, S, 48);                      \
    ASYNC_LD16(L, V, S, 64);  ASYNC_LD16(L, V, S, 80);                      \
    ASYNC_LD16(L, V, S, 96);  ASYNC_LD16(L, V, S, 112);                     \
    ASYNC_LD16(L, V, S, 128); ASYNC_LD16(L, V, S, 144);                     \
    ASYNC_LD16(L, V, S, 160); ASYNC_LD16(L, V, S, 176);                     \
    ASYNC_LD16(L, V, S, 192); ASYNC_LD16(L, V, S, 208);                     \
    ASYNC_LD16(L, V, S, 224); ASYNC_LD16(L, V, S, 240)

#define ASYNC_TILE8(L, V, S)                                                \
    ASYNC_LD16(L, V, S, 0);   ASYNC_LD16(L, V, S, 16);                      \
    ASYNC_LD16(L, V, S, 32);  ASYNC_LD16(L, V, S, 48);                      \
    ASYNC_LD16(L, V, S, 64);  ASYNC_LD16(L, V, S, 80);                      \
    ASYNC_LD16(L, V, S, 96);  ASYNC_LD16(L, V, S, 112)

#define WAIT_ASYNC0() asm volatile("s_wait_asynccnt 0x0" ::: "memory")

// low 32 bits of a generic pointer to __shared__ == DS byte address
static __device__ __forceinline__ unsigned lds_addr32(const void* p) {
    return (unsigned)(size_t)p;
}

// ---------------- preprocess: L2-normalize rows of (rows x 256) -> bf16 -----
__global__ __launch_bounds__(256) void csa_norm_bf16(const float* __restrict__ src,
                                                     __bf16* __restrict__ dst,
                                                     int rows) {
    const int wave = blockIdx.x * 8 + (threadIdx.x >> 5);
    const int lane = threadIdx.x & 31;
    if (wave >= rows) return;
    const float* r = src + (size_t)wave * DDIM + lane * 8;
    v4f a = *(const v4f*)r;
    v4f b = *(const v4f*)(r + 4);
    float ss = a.x*a.x + a.y*a.y + a.z*a.z + a.w*a.w
             + b.x*b.x + b.y*b.y + b.z*b.z + b.w*b.w;
    ss += __shfl_xor(ss, 1, 32);
    ss += __shfl_xor(ss, 2, 32);
    ss += __shfl_xor(ss, 4, 32);
    ss += __shfl_xor(ss, 8, 32);
    ss += __shfl_xor(ss, 16, 32);
    const float s = 1.0f / fmaxf(sqrtf(ss), 1e-12f);
    v8bf o;
    o[0] = to_bf16(a.x * s); o[1] = to_bf16(a.y * s);
    o[2] = to_bf16(a.z * s); o[3] = to_bf16(a.w * s);
    o[4] = to_bf16(b.x * s); o[5] = to_bf16(b.y * s);
    o[6] = to_bf16(b.z * s); o[7] = to_bf16(b.w * s);
    *(v8bf*)(dst + (size_t)wave * DDIM + lane * 8) = o;
}

// ---------------- preprocess: V[b][n][d] f32 -> Vt[b][d][n] bf16 ------------
__global__ __launch_bounds__(256) void csa_transpose_v(const float* __restrict__ v,
                                                       __bf16* __restrict__ vt) {
    const int idx = blockIdx.x * 256 + threadIdx.x;     // B*NK*32 total
    const int dc = idx & 31;
    const int n  = (idx >> 5) & (NK - 1);
    const int b  = idx >> 17;
    const int d0 = dc * 8;
    const float* src = v + ((size_t)(b * NK + n)) * DDIM + d0;
    v4f a = *(const v4f*)src;
    v4f c = *(const v4f*)(src + 4);
    float vals[8] = {a.x, a.y, a.z, a.w, c.x, c.y, c.z, c.w};
#pragma unroll
    for (int i = 0; i < 8; ++i)
        vt[((size_t)(b * DDIM + d0 + i)) * NK + n] = to_bf16(vals[i]);
}

// ---------------- flash attention over bf16 Qn/Kn/Vt ------------------------
__global__ __launch_bounds__(128) void csa_attention(const __bf16* __restrict__ Qn,
                                                     const __bf16* __restrict__ Kn,
                                                     const __bf16* __restrict__ Vt,
                                                     float* __restrict__ out) {
    // double-buffered tiles + dedicated P staging (~147 KB of gfx1250's 320 KB)
    __shared__ __attribute__((aligned(16))) __bf16 lds_kb[2][BM * KSTRIDE];
    __shared__ __attribute__((aligned(16))) __bf16 lds_vb[2][DDIM * VSTRIDE];
    __shared__ __attribute__((aligned(16))) __bf16 lds_p[4 * 16 * PSTRIDE];

    const int tid  = threadIdx.x;
    const int wave = tid >> 5;
    const int lane = tid & 31;
    const int lh   = (lane >> 4) & 1;   // lane half
    const int lm   = lane & 15;
    const int b    = blockIdx.y;
    const int q0   = blockIdx.x * BM;
    const int qrow = q0 + wave * 16 + lm;

    const unsigned long long kbase =
        (unsigned long long)(size_t)(Kn + (size_t)b * NK * DDIM);
    const unsigned long long vbase =
        (unsigned long long)(size_t)(Vt + (size_t)b * DDIM * NK);

    // per-thread cooperative tile-load coordinates
    const int kr = tid >> 1;                 // K tile row   (0..63)
    const int dh = (tid & 1) * 128;          // K tile d-half
    const int vd0 = tid;                     // Vt rows vd0 and vd0+128

    // issue one BN-wide K/Vt tile into buffer `buf` via async-to-LDS
    auto issue_tile = [&](int buf, int k0) {
        const unsigned klds = lds_addr32(&lds_kb[buf][kr * KSTRIDE + dh]);
        const unsigned kvo  = (unsigned)(((k0 + kr) * DDIM + dh) * 2);
        ASYNC_TILE16(klds, kvo, kbase);
#pragma unroll
        for (int rep = 0; rep < 2; ++rep) {
            const int d = vd0 + rep * 128;
            const unsigned vlds = lds_addr32(&lds_vb[buf][d * VSTRIDE]);
            const unsigned vvo  = (unsigned)((d * NK + k0) * 2);
            ASYNC_TILE8(vlds, vvo, vbase);
        }
    };

    // --- persistent Q A-fragments (16 rows x 256, 8 chunks of K=32) ---------
    v16bf qf[8];
    {
        const __bf16* qb = Qn + ((size_t)(b * NQ + qrow)) * DDIM;
#pragma unroll
        for (int c = 0; c < 8; ++c) {
            const int d0 = c * 32 + lh * 8;
            qf[c] = cat8(*(const v8bf*)(qb + d0), *(const v8bf*)(qb + d0 + 16));
        }
    }

    v8f o[16];
#pragma unroll
    for (int t = 0; t < 16; ++t)
#pragma unroll
        for (int e = 0; e < 8; ++e) o[t][e] = 0.0f;

    float mrow[8], lrow[8];
#pragma unroll
    for (int r = 0; r < 8; ++r) { mrow[r] = -1e30f; lrow[r] = 0.0f; }

    __bf16* pbuf = lds_p + wave * (16 * PSTRIDE);   // per-wave P slice

    issue_tile(0, 0);                               // prologue prefetch
    int cur = 0;

    for (int k0 = 0; k0 < NK; k0 += BN) {
        WAIT_ASYNC0();       // this wave's async writes for tile `cur` landed
        __syncthreads();     // all waves' tiles landed; prev buffer reads done

        if (k0 + BN < NK) issue_tile(cur ^ 1, k0 + BN);   // prefetch next

        const __bf16* kt = lds_kb[cur];
        const __bf16* vt = lds_vb[cur];

        // ---- S = Q * K^T : 4 N-subtiles x 8 K-steps ------------------------
        v8f s[4];
#pragma unroll
        for (int j = 0; j < 4; ++j) {
            v8f acc;
#pragma unroll
            for (int e = 0; e < 8; ++e) acc[e] = 0.0f;
#pragma unroll
            for (int c = 0; c < 8; ++c) {
                const __bf16* kp = kt + (j * 16 + lm) * KSTRIDE + c * 32 + lh * 8;
                v16bf bf = cat8(*(const v8bf*)kp, *(const v8bf*)(kp + 16));
                acc = __builtin_amdgcn_wmma_f32_16x16x32_bf16(
                        false, qf[c], false, bf, (short)0, acc, false, false);
            }
            s[j] = acc;
        }

        // ---- online softmax ------------------------------------------------
        float alpha[8];
#pragma unroll
        for (int r = 0; r < 8; ++r) {
            float v = fmaxf(fmaxf(s[0][r], s[1][r]), fmaxf(s[2][r], s[3][r])) * SCALE;
            v = rmax16(v);
            const float mnew = fmaxf(mrow[r], v);
            alpha[r] = __expf(mrow[r] - mnew);
            mrow[r] = mnew;
        }
#pragma unroll
        for (int j = 0; j < 4; ++j)
#pragma unroll
            for (int r = 0; r < 8; ++r)
                s[j][r] = __expf(s[j][r] * SCALE - mrow[r]);
#pragma unroll
        for (int r = 0; r < 8; ++r) {
            const float ls = rsum16(s[0][r] + s[1][r] + s[2][r] + s[3][r]);
            lrow[r] = lrow[r] * alpha[r] + ls;
        }
#pragma unroll
        for (int t = 0; t < 16; ++t)
#pragma unroll
            for (int r = 0; r < 8; ++r) o[t][r] *= alpha[r];

        // ---- P (C-layout regs) -> private LDS slice, row-major 16x64 -------
#pragma unroll
        for (int j = 0; j < 4; ++j)
#pragma unroll
            for (int r = 0; r < 8; ++r)
                pbuf[(r + 8 * lh) * PSTRIDE + j * 16 + lm] = to_bf16(s[j][r]);

        // ---- O += P * V : 2 K-steps x 16 D-subtiles ------------------------
#pragma unroll
        for (int kk = 0; kk < 2; ++kk) {
            const __bf16* pp = pbuf + lm * PSTRIDE + kk * 32 + lh * 8;
            v16bf pf = cat8(*(const v8bf*)pp, *(const v8bf*)(pp + 16));
#pragma unroll
            for (int t = 0; t < 16; ++t) {
                const __bf16* vp = vt + (t * 16 + lm) * VSTRIDE + kk * 32 + lh * 8;
                v16bf vf = cat8(*(const v8bf*)vp, *(const v8bf*)(vp + 16));
                o[t] = __builtin_amdgcn_wmma_f32_16x16x32_bf16(
                        false, pf, false, vf, (short)0, o[t], false, false);
            }
        }

        cur ^= 1;
    }

    // ---- epilogue: O / l -> global ----------------------------------------
    float inv[8];
#pragma unroll
    for (int r = 0; r < 8; ++r) inv[r] = 1.0f / lrow[r];
    float* ob = out + ((size_t)(b * NQ + q0 + wave * 16)) * DDIM;
#pragma unroll
    for (int t = 0; t < 16; ++t)
#pragma unroll
        for (int r = 0; r < 8; ++r)
            ob[(size_t)(r + 8 * lh) * DDIM + t * 16 + lm] = o[t][r] * inv[r];
}

// ---------------------------------------------------------------------------
extern "C" void kernel_launch(void* const* d_in, const int* in_sizes, int n_in,
                              void* d_out, int out_size, void* d_ws, size_t ws_size,
                              hipStream_t stream) {
    (void)in_sizes; (void)n_in; (void)out_size; (void)ws_size;
    const float* q = (const float*)d_in[0];
    const float* k = (const float*)d_in[1];
    const float* v = (const float*)d_in[2];
    float* out = (float*)d_out;

    __bf16* qn = (__bf16*)d_ws;                                  // 8 MB
    __bf16* kn = qn + (size_t)BATCH * NQ * DDIM;                 // 8 MB
    __bf16* vt = kn + (size_t)BATCH * NK * DDIM;                 // 8 MB

    const int rows = BATCH * NQ;                 // 16384
    csa_norm_bf16<<<rows / 8, 256, 0, stream>>>(q, qn, rows);
    csa_norm_bf16<<<rows / 8, 256, 0, stream>>>(k, kn, rows);
    csa_transpose_v<<<(BATCH * NK * 32) / 256, 256, 0, stream>>>(v, vt);

    dim3 grid(NQ / BM, BATCH);
    csa_attention<<<grid, 128, 0, stream>>>(qn, kn, vt, out);
}
